// SNet_dfver2_6708738916439
// MI455X (gfx1250) — compile-verified, matrix operand order checked by reading, not run on hardware
//
#include <hip/hip_runtime.h>
#include <hip/hip_bf16.h>

// ---------------------------------------------------------------------------
// Types for CDNA5 WMMA (wave32, 16x16x32 bf16 -> f32)
// ---------------------------------------------------------------------------
typedef __attribute__((ext_vector_type(16))) __bf16 v16bf;
typedef __attribute__((ext_vector_type(8)))  float  v8f;
typedef unsigned short u16;

struct alignas(16) U16B { unsigned int a, b, c, d; };
struct U32B { U16B lo, hi; };   // 32 bytes == v16bf

__device__ inline u16 f2bf(float f) {
    unsigned int u = __builtin_bit_cast(unsigned int, f);
    u += 0x7FFFu + ((u >> 16) & 1u);          // round-to-nearest-even
    return (u16)(u >> 16);
}
__device__ inline float bf2f(u16 h) {
    unsigned int u = ((unsigned int)h) << 16;
    return __builtin_bit_cast(float, u);
}
__device__ inline float bflo(unsigned int w) {
    return __builtin_bit_cast(float, w << 16);
}
__device__ inline float bfhi(unsigned int w) {
    return __builtin_bit_cast(float, w & 0xFFFF0000u);
}

// Fixed problem dims: N=16, H=W=256 -> pixel index p = (img<<16)|(y<<8)|x
#define IMG_OF(p)  ((p) >> 16)
#define Y_OF(p)    (((p) >> 8) & 255)
#define X_OF(p)    ((p) & 255)

// ---------------------------------------------------------------------------
// Weight packing: OIHW f32 [Cout][64][3][3] -> bf16 fragments
// layout [t(9)][cg(2)][ntile(NT)][lane(32)][half(16)]
// lane<16 : halves 0..7 -> K 0..7,  halves 8..15 -> K 16..23
// lane>=16: halves 0..7 -> K 8..15, halves 8..15 -> K 24..31
// ---------------------------------------------------------------------------
__global__ void pack_weights(const float* __restrict__ w, u16* __restrict__ out,
                             int NT, int coutReal) {
    int idx = blockIdx.x * blockDim.x + threadIdx.x;
    int total = 9 * 2 * NT * 32 * 16;
    if (idx >= total) return;
    int h    = idx & 15;
    int lane = (idx >> 4) & 31;
    int rest = idx >> 9;
    int n  = rest % NT; rest /= NT;
    int cg = rest & 1;
    int t  = rest >> 1;
    int cout = n * 16 + (lane & 15);
    int k    = ((lane >> 4) << 3) + (h & 7) + ((h >> 3) << 4);
    int cin  = cg * 32 + k;
    float val = (cout < coutReal) ? w[((size_t)cout * 64 + cin) * 9 + t] : 0.0f;
    out[idx] = f2bf(val);
}

__global__ void pad_bias(const float* __restrict__ b, float* __restrict__ out,
                         int realC, int totC) {
    int i = blockIdx.x * blockDim.x + threadIdx.x;
    if (i < totC) out[i] = (i < realC) ? b[i] : 0.0f;
}

// ---------------------------------------------------------------------------
// Implicit-GEMM 3x3 SAME conv, Cin=64 (bf16 NHWC) -> Cout=NT*16 (bf16 NHWC)
// One wave: 16 pixels (M) x NT*16 couts (N), K = 9*64 in 18 steps of 32.
// Fast path (interior tiles): fully unrolled, immediate-offset b128 loads.
// Slow path (border tiles): clamped addresses + value masking.
// ---------------------------------------------------------------------------
template<int NT, bool RELU>
__global__ __launch_bounds__(256)
void conv3x3_wmma(const u16* __restrict__ in,      // [p][64] bf16
                  const u16* __restrict__ wpack,   // packed fragments
                  const float* __restrict__ biasPad, // [NT*16] f32, zero padded
                  u16* __restrict__ out,           // [p][NT*16] bf16
                  int tiles) {
    constexpr int NTC = NT * 16;
    const int lane   = threadIdx.x & 31;
    const int waveId = blockIdx.x * 8 + (threadIdx.x >> 5);
    const int stride = gridDim.x * 8;

    const int mrow  = lane & 15;            // pixel-in-tile / D column id
    const int kbase = (lane >> 4) << 3;     // 0 or 8 (K sub-offset)
    const int rbase = (lane >> 4) << 3;     // D rows 0..7 or 8..15

    const u16* wlane = wpack + (size_t)lane * 16;   // + ((t*2+cg)*NT+n)*512

    for (int tile = waveId; tile < tiles; tile += stride) {
        const int pbase = tile << 4;
        const int img = IMG_OF(pbase);
        const int y   = Y_OF(pbase);
        const int x0  = X_OF(pbase);
        const int pix = pbase + mrow;       // this lane's pixel

        v8f acc[NT];
#pragma unroll
        for (int n = 0; n < NT; ++n)
#pragma unroll
            for (int i = 0; i < 8; ++i) acc[n][i] = 0.0f;

        if (y >= 1 && y <= 254 && x0 != 0 && x0 != 240) {
            // ---------------- interior fast path ----------------
            const u16* ap = in + (size_t)pix * 64 + kbase;
#pragma unroll
            for (int t = 0; t < 9; ++t) {
                const int doff = ((t / 3 - 1) * 256 + (t % 3 - 1)) * 64;
#pragma unroll
                for (int cg = 0; cg < 2; ++cg) {
                    U32B ta;
                    ta.lo = *(const U16B*)(ap + doff + cg * 32);
                    ta.hi = *(const U16B*)(ap + doff + cg * 32 + 16);
                    const v16bf afrag = __builtin_bit_cast(v16bf, ta);
#pragma unroll
                    for (int n = 0; n < NT; ++n) {
                        const u16* wb = wlane + (size_t)((t * 2 + cg) * NT + n) * 512;
                        U32B tb;
                        tb.lo = *(const U16B*)(wb);
                        tb.hi = *(const U16B*)(wb + 8);
                        const v16bf bfrag = __builtin_bit_cast(v16bf, tb);
                        acc[n] = __builtin_amdgcn_wmma_f32_16x16x32_bf16(
                            false, afrag, false, bfrag, (short)0, acc[n], false, false);
                    }
                }
            }
        } else {
            // ---------------- border slow path ----------------
            const int x = x0 + mrow;
            for (int t = 0; t < 9; ++t) {
                const int yy = y + t / 3 - 1;
                const int xx = x + t % 3 - 1;
                const bool ok = (yy >= 0) && (yy <= 255) && (xx >= 0) && (xx <= 255);
                const int yc = yy < 0 ? 0 : (yy > 255 ? 255 : yy);
                const int xc = xx < 0 ? 0 : (xx > 255 ? 255 : xx);
                const unsigned msk = ok ? 0xFFFFFFFFu : 0u;
                const u16* ap = in + (size_t)(((img << 8) + yc) * 256 + xc) * 64 + kbase;
#pragma unroll
                for (int cg = 0; cg < 2; ++cg) {
                    U32B ta;
                    ta.lo = *(const U16B*)(ap + cg * 32);
                    ta.hi = *(const U16B*)(ap + cg * 32 + 16);
                    ta.lo.a &= msk; ta.lo.b &= msk; ta.lo.c &= msk; ta.lo.d &= msk;
                    ta.hi.a &= msk; ta.hi.b &= msk; ta.hi.c &= msk; ta.hi.d &= msk;
                    const v16bf afrag = __builtin_bit_cast(v16bf, ta);
#pragma unroll
                    for (int n = 0; n < NT; ++n) {
                        const u16* wb = wlane + (size_t)((t * 2 + cg) * NT + n) * 512;
                        U32B tb;
                        tb.lo = *(const U16B*)(wb);
                        tb.hi = *(const U16B*)(wb + 8);
                        const v16bf bfrag = __builtin_bit_cast(v16bf, tb);
                        acc[n] = __builtin_amdgcn_wmma_f32_16x16x32_bf16(
                            false, afrag, false, bfrag, (short)0, acc[n], false, false);
                    }
                }
            }
        }

        // ---------------- epilogue: bias (+relu), store bf16 NHWC ----------------
        u16* op = out + (size_t)(pbase + rbase) * NTC + mrow;
#pragma unroll
        for (int n = 0; n < NT; ++n) {
            const float bv = biasPad[n * 16 + mrow];
#pragma unroll
            for (int r = 0; r < 8; ++r) {
                float v = acc[n][r] + bv;
                if (RELU) v = v > 0.0f ? v : 0.0f;
                op[(size_t)r * NTC + n * 16] = f2bf(v);
            }
        }
    }
}

// ---------------------------------------------------------------------------
// Small direct-conv / pointwise kernels (f32 VALU; <5% of total FLOPs)
// ---------------------------------------------------------------------------
__global__ void conv1_1to64(const float* __restrict__ x, const float* __restrict__ w,
                            const float* __restrict__ b, u16* __restrict__ out,
                            int total) {
    int p = blockIdx.x * blockDim.x + threadIdx.x;
    if (p >= total) return;
    int img = IMG_OF(p), y = Y_OF(p), xx = X_OF(p);
    float v[9];
#pragma unroll
    for (int t = 0; t < 9; ++t) {
        int yy = y + t / 3 - 1, xc = xx + t % 3 - 1;
        bool ok = (yy >= 0 && yy <= 255 && xc >= 0 && xc <= 255);
        v[t] = ok ? x[(size_t)(((img << 8) + yy) << 8) + xc] : 0.0f;
    }
    for (int co = 0; co < 64; ++co) {
        float s = b[co];
#pragma unroll
        for (int t = 0; t < 9; ++t) s += w[co * 9 + t] * v[t];
        s = s > 0.0f ? s : 0.0f;
        out[(size_t)p * 64 + co] = f2bf(s);
    }
}

__global__ void softmax81(u16* __restrict__ buf, int total) {
    int p = blockIdx.x * blockDim.x + threadIdx.x;
    if (p >= total) return;
    u16* q = buf + (size_t)p * 96;
    float m = -1e30f;
    for (int i = 0; i < 81; ++i) { float v = bf2f(q[i]); m = v > m ? v : m; }
    float s = 0.0f;
    for (int i = 0; i < 81; ++i) s += __expf(bf2f(q[i]) - m);
    float inv = 1.0f / s;
    for (int i = 0; i < 81; ++i) q[i] = f2bf(__expf(bf2f(q[i]) - m) * inv);
}

__global__ void apply_filters(const float* __restrict__ x, const float* __restrict__ g,
                              const u16* __restrict__ fx, const u16* __restrict__ fg,
                              float* __restrict__ pair, int total) {
    int p = blockIdx.x * blockDim.x + threadIdx.x;
    if (p >= total) return;
    int img = IMG_OF(p), y = Y_OF(p), xx = X_OF(p);
    const u16* fxp = fx + (size_t)p * 96;
    const u16* fgp = fg + (size_t)p * 96;
    float sx = 0.0f, sg = 0.0f;
    for (int t = 0; t < 81; ++t) {
        int yy = y + t / 9 - 4, xc = xx + t % 9 - 4;
        bool ok = (yy >= 0 && yy <= 255 && xc >= 0 && xc <= 255);
        int yc = yy < 0 ? 0 : (yy > 255 ? 255 : yy);
        int xcc = xc < 0 ? 0 : (xc > 255 ? 255 : xc);
        size_t off = (size_t)(((img << 8) + yc) << 8) + xcc;
        float xv = ok ? x[off] : 0.0f;
        float gv = ok ? g[off] : 0.0f;
        sx += xv * bf2f(fxp[t]);
        sg += gv * bf2f(fgp[t]);
    }
    pair[(size_t)p * 2 + 0] = sx;
    pair[(size_t)p * 2 + 1] = sg;
}

__global__ void conv1_2to64(const float* __restrict__ pair, const float* __restrict__ w,
                            const float* __restrict__ b, u16* __restrict__ out,
                            int total) {
    int p = blockIdx.x * blockDim.x + threadIdx.x;
    if (p >= total) return;
    int img = IMG_OF(p), y = Y_OF(p), xx = X_OF(p);
    float v[18];
#pragma unroll
    for (int t = 0; t < 9; ++t) {
        int yy = y + t / 3 - 1, xc = xx + t % 3 - 1;
        bool ok = (yy >= 0 && yy <= 255 && xc >= 0 && xc <= 255);
        size_t q = (size_t)((((img << 8) + (yy & 255)) << 8) + (xc & 255)) * 2;
        v[t * 2 + 0] = ok ? pair[q + 0] : 0.0f;
        v[t * 2 + 1] = ok ? pair[q + 1] : 0.0f;
    }
    for (int co = 0; co < 64; ++co) {
        float s = b[co];
#pragma unroll
        for (int t = 0; t < 9; ++t) {
            s += w[(co * 2 + 0) * 9 + t] * v[t * 2 + 0];
            s += w[(co * 2 + 1) * 9 + t] * v[t * 2 + 1];
        }
        s = s > 0.0f ? s : 0.0f;
        out[(size_t)p * 64 + co] = f2bf(s);
    }
}

__global__ void conv_64to1(const u16* __restrict__ in, const float* __restrict__ w,
                           const float* __restrict__ b, float* __restrict__ out,
                           int total) {
    int p = blockIdx.x * blockDim.x + threadIdx.x;
    if (p >= total) return;
    int img = IMG_OF(p), y = Y_OF(p), xx = X_OF(p);
    float s = b[0];
    for (int t = 0; t < 9; ++t) {
        int yy = y + t / 3 - 1, xc = xx + t % 3 - 1;
        if (yy < 0 || yy > 255 || xc < 0 || xc > 255) continue;
        const u16* ap = in + (size_t)((((img << 8) + yy) << 8) + xc) * 64;
#pragma unroll
        for (int c8 = 0; c8 < 64; c8 += 8) {
            U16B vv = *(const U16B*)(ap + c8);
            s += bflo(vv.a) * w[(c8 + 0) * 9 + t] + bfhi(vv.a) * w[(c8 + 1) * 9 + t];
            s += bflo(vv.b) * w[(c8 + 2) * 9 + t] + bfhi(vv.b) * w[(c8 + 3) * 9 + t];
            s += bflo(vv.c) * w[(c8 + 4) * 9 + t] + bfhi(vv.c) * w[(c8 + 5) * 9 + t];
            s += bflo(vv.d) * w[(c8 + 6) * 9 + t] + bfhi(vv.d) * w[(c8 + 7) * 9 + t];
        }
    }
    out[p] = s;
}

// ---------------------------------------------------------------------------
// Orchestration
// ---------------------------------------------------------------------------
extern "C" void kernel_launch(void* const* d_in, const int* in_sizes, int n_in,
                              void* d_out, int out_size, void* d_ws, size_t ws_size,
                              hipStream_t stream) {
    (void)in_sizes; (void)n_in; (void)out_size; (void)ws_size;
    const size_t P = (size_t)16 * 256 * 256;            // 1,048,576 pixels
    const int tiles = (int)(P >> 4);                    // 65,536 16-pixel tiles

    const float* x     = (const float*)d_in[0];
    const float* g     = (const float*)d_in[1];
    const float* fx_w1 = (const float*)d_in[2];  const float* fx_b1 = (const float*)d_in[3];
    const float* fx_w2 = (const float*)d_in[4];  const float* fx_b2 = (const float*)d_in[5];
    const float* fx_w3 = (const float*)d_in[6];  const float* fx_b3 = (const float*)d_in[7];
    const float* fg_w1 = (const float*)d_in[8];  const float* fg_b1 = (const float*)d_in[9];
    const float* fg_w2 = (const float*)d_in[10]; const float* fg_b2 = (const float*)d_in[11];
    const float* fg_w3 = (const float*)d_in[12]; const float* fg_b3 = (const float*)d_in[13];
    const float* c1_w  = (const float*)d_in[14]; const float* c1_b  = (const float*)d_in[15];
    const float* c2_w  = (const float*)d_in[16]; const float* c2_b  = (const float*)d_in[17];
    const float* c3_w  = (const float*)d_in[18]; const float* c3_b  = (const float*)d_in[19];
    float* out = (float*)d_out;

    // Workspace carve-out
    char* base = (char*)d_ws;
    size_t off = 0;
    auto carve = [&](size_t bytes) {
        void* p = base + off;
        off = (off + bytes + 255) & ~(size_t)255;
        return p;
    };
    u16*   buf0    = (u16*)  carve(P * 64 * 2);   // 64-ch activations (stage A)
    u16*   buf1    = (u16*)  carve(P * 64 * 2);   // 64-ch activations (stage B)
    u16*   bufFx   = (u16*)  carve(P * 96 * 2);   // 96-ch logits / filters for x
    u16*   bufFg   = (u16*)  carve(P * 96 * 2);   // 96-ch logits / filters for g
    float* bufPair = (float*)carve(P * 2 * 4);    // [x_, g_] per pixel
    u16*   wp2x    = (u16*)  carve(9 * 2 * 4 * 512 * 2);
    u16*   wp3x    = (u16*)  carve(9 * 2 * 6 * 512 * 2);
    u16*   wp2g    = (u16*)  carve(9 * 2 * 4 * 512 * 2);
    u16*   wp3g    = (u16*)  carve(9 * 2 * 6 * 512 * 2);
    u16*   wpc2    = (u16*)  carve(9 * 2 * 4 * 512 * 2);
    float* pb2x    = (float*)carve(96 * 4);
    float* pb3x    = (float*)carve(96 * 4);
    float* pb2g    = (float*)carve(96 * 4);
    float* pb3g    = (float*)carve(96 * 4);
    float* pbc2    = (float*)carve(96 * 4);

    const int TPB = 256;
    const int pixBlocks  = (int)((P + TPB - 1) / TPB);  // 4096
    const int tileBlocks = (tiles + 7) / 8;             // 8192 (8 waves/block)
    const int pk4 = (9 * 2 * 4 * 512 + TPB - 1) / TPB;
    const int pk6 = (9 * 2 * 6 * 512 + TPB - 1) / TPB;

    // 0) pack WMMA weight fragments + padded biases
    pack_weights<<<pk4, TPB, 0, stream>>>(fx_w2, wp2x, 4, 64);
    pack_weights<<<pk6, TPB, 0, stream>>>(fx_w3, wp3x, 6, 81);
    pack_weights<<<pk4, TPB, 0, stream>>>(fg_w2, wp2g, 4, 64);
    pack_weights<<<pk6, TPB, 0, stream>>>(fg_w3, wp3g, 6, 81);
    pack_weights<<<pk4, TPB, 0, stream>>>(c2_w,  wpc2, 4, 64);
    pad_bias<<<1, 96, 0, stream>>>(fx_b2, pb2x, 64, 64);
    pad_bias<<<1, 96, 0, stream>>>(fx_b3, pb3x, 81, 96);
    pad_bias<<<1, 96, 0, stream>>>(fg_b2, pb2g, 64, 64);
    pad_bias<<<1, 96, 0, stream>>>(fg_b3, pb3g, 81, 96);
    pad_bias<<<1, 96, 0, stream>>>(c2_b,  pbc2, 64, 64);

    // 1) dynamic-filter head for x
    conv1_1to64<<<pixBlocks, TPB, 0, stream>>>(x, fx_w1, fx_b1, buf0, (int)P);
    conv3x3_wmma<4, true ><<<tileBlocks, TPB, 0, stream>>>(buf0, wp2x, pb2x, buf1, tiles);
    conv3x3_wmma<6, false><<<tileBlocks, TPB, 0, stream>>>(buf1, wp3x, pb3x, bufFx, tiles);
    softmax81<<<pixBlocks, TPB, 0, stream>>>(bufFx, (int)P);

    // 2) dynamic-filter head for g
    conv1_1to64<<<pixBlocks, TPB, 0, stream>>>(g, fg_w1, fg_b1, buf0, (int)P);
    conv3x3_wmma<4, true ><<<tileBlocks, TPB, 0, stream>>>(buf0, wp2g, pb2g, buf1, tiles);
    conv3x3_wmma<6, false><<<tileBlocks, TPB, 0, stream>>>(buf1, wp3g, pb3g, bufFg, tiles);
    softmax81<<<pixBlocks, TPB, 0, stream>>>(bufFg, (int)P);

    // 3) pixel-adaptive 9x9 filtering
    apply_filters<<<pixBlocks, TPB, 0, stream>>>(x, g, bufFx, bufFg, bufPair, (int)P);

    // 4) fusion head
    conv1_2to64<<<pixBlocks, TPB, 0, stream>>>(bufPair, c1_w, c1_b, buf0, (int)P);
    conv3x3_wmma<4, true ><<<tileBlocks, TPB, 0, stream>>>(buf0, wpc2, pbc2, buf1, tiles);
    conv_64to1<<<pixBlocks, TPB, 0, stream>>>(buf1, c3_w, c3_b, out, (int)P);
}
